// MipNerfModel_1142461301252
// MI455X (gfx1250) — matrix operand order repaired and checked by reading, not compiled, not run on hardware
//
#include <hip/hip_runtime.h>
#include <hip/hip_bf16.h>
#include <math.h>

typedef __attribute__((ext_vector_type(16))) _Float16 v16h;
typedef __attribute__((ext_vector_type(8)))  _Float16 v8h;
typedef __attribute__((ext_vector_type(8)))  float    v8f;

static constexpr int NB    = 1024;     // rays
static constexpr int NS    = 128;      // samples per ray
static constexpr int MTOT  = NB * NS;  // 131072
static constexpr int CH    = 32768;    // samples per chunk (256 rays)
static constexpr int NCHUNK = MTOT / CH;
static constexpr int LDACT = 352;      // activation row stride (256 x + 96 ipe)

// ---------------------------------------------------------------------------
// Weight transpose + fp16 convert:  W[K][N] fp32  ->  Wt[N][KPAD] fp16 (0-pad)
// ---------------------------------------------------------------------------
__global__ void mnf_transpose_w(const float* __restrict__ W, _Float16* __restrict__ Wt,
                                int K, int N, int KPAD) {
    int i = blockIdx.x * blockDim.x + threadIdx.x;
    if (i >= N * KPAD) return;
    int n = i / KPAD, k = i - n * KPAD;
    float v = (k < K) ? W[(size_t)k * N + n] : 0.0f;
    Wt[i] = (_Float16)v;
}

// ---------------------------------------------------------------------------
// t_vals init (level 0): near + (far-near)*i/128, per ray, 129 values
// ---------------------------------------------------------------------------
__global__ void mnf_tvals_init(const float* __restrict__ nearp, const float* __restrict__ farp,
                               float* __restrict__ tvals) {
    int i = blockIdx.x * blockDim.x + threadIdx.x;
    if (i >= NB * (NS + 1)) return;
    int b = i / (NS + 1), s = i - b * (NS + 1);
    tvals[i] = nearp[b] + (farp[b] - nearp[b]) * ((float)s / (float)NS);
}

// ---------------------------------------------------------------------------
// cast_rays + integrated_pos_enc for one chunk of samples.
// Writes 96 fp16 IPE features into columns [256,352) of both ping-pong buffers.
// ---------------------------------------------------------------------------
__global__ void mnf_prep(const float* __restrict__ tvals, const float* __restrict__ origins,
                         const float* __restrict__ dirs, const float* __restrict__ radii,
                         _Float16* __restrict__ Abuf, _Float16* __restrict__ Bbuf, int mbase) {
    int r = blockIdx.x * blockDim.x + threadIdx.x;
    if (r >= CH) return;
    int m = mbase + r;
    int b = m >> 7, s = m & 127;

    float t0 = tvals[b * (NS + 1) + s];
    float t1 = tvals[b * (NS + 1) + s + 1];
    float mu = 0.5f * (t0 + t1), hw = 0.5f * (t1 - t0);
    float mu2 = mu * mu, hw2 = hw * hw, hw4 = hw2 * hw2;
    float denom = 3.0f * mu2 + hw2;
    float t_mean = mu + 2.0f * mu * hw2 / denom;
    float t_var  = hw2 * (1.0f / 3.0f) - (4.0f / 15.0f) * (hw4 * (12.0f * mu2 - hw2)) / (denom * denom);
    float rad = radii[b];
    float r_var = rad * rad * (mu2 * 0.25f + (5.0f / 12.0f) * hw2 - (4.0f / 15.0f) * hw4 / denom);

    float d0 = dirs[b * 3 + 0], d1 = dirs[b * 3 + 1], d2 = dirs[b * 3 + 2];
    float o0 = origins[b * 3 + 0], o1 = origins[b * 3 + 1], o2 = origins[b * 3 + 2];
    float dsq = fmaxf(1e-10f, d0 * d0 + d1 * d1 + d2 * d2);

    float mean[3], cov[3];
    float dd[3] = {d0, d1, d2};
    float oo[3] = {o0, o1, o2};
    #pragma unroll
    for (int c = 0; c < 3; ++c) {
        mean[c] = oo[c] + dd[c] * t_mean;
        float dout = dd[c] * dd[c];
        cov[c] = t_var * dout + r_var * (1.0f - dout / dsq);
    }

    _Float16* arow = Abuf + (size_t)r * LDACT + 256;
    _Float16* brow = Bbuf + (size_t)r * LDACT + 256;
    float scale = 1.0f;
    for (int j = 0; j < 16; ++j) {
        #pragma unroll
        for (int c = 0; c < 3; ++c) {
            float y  = mean[c] * scale;
            float yv = cov[c] * scale * scale;
            float e  = expf(-0.5f * yv);
            _Float16 fs = (_Float16)(e * sinf(y));
            _Float16 fc = (_Float16)(e * cosf(y));
            arow[j * 3 + c]      = fs;  brow[j * 3 + c]      = fs;
            arow[48 + j * 3 + c] = fc;  brow[48 + j * 3 + c] = fc;
        }
        scale *= 2.0f;
    }
}

// ---------------------------------------------------------------------------
// View-direction positional encoding -> columns [256,288) of Bbuf (27 + 5 pad)
// ---------------------------------------------------------------------------
__global__ void mnf_vdenc(const float* __restrict__ viewdirs, _Float16* __restrict__ Bbuf,
                          int mbase) {
    int r = blockIdx.x * blockDim.x + threadIdx.x;
    if (r >= CH) return;
    int b = (mbase + r) >> 7;
    float v[3] = {viewdirs[b * 3], viewdirs[b * 3 + 1], viewdirs[b * 3 + 2]};
    _Float16* row = Bbuf + (size_t)r * LDACT + 256;
    row[0] = (_Float16)v[0]; row[1] = (_Float16)v[1]; row[2] = (_Float16)v[2];
    float scale = 1.0f;
    for (int j = 0; j < 4; ++j) {
        #pragma unroll
        for (int c = 0; c < 3; ++c) {
            float x = v[c] * scale;
            row[3 + j * 3 + c]      = (_Float16)sinf(x);
            row[3 + 12 + j * 3 + c] = (_Float16)cosf(x);
        }
        scale *= 2.0f;
    }
    #pragma unroll
    for (int j = 27; j < 32; ++j) row[j] = (_Float16)0.0f;
}

// ---------------------------------------------------------------------------
// WMMA GEMM:  Y[64-row tiles][N] = act(X[M][KPAD] @ Wt^T + bias)
//   Wt stored [N][KPAD] fp16 so each lane's B-fragment is contiguous.
//   Block: 256 threads = 8 waves; M-tile 64 rows staged in LDS; wave w owns
//   columns [w*16*NFRAG, (w+1)*16*NFRAG).  K fully unrolled (KPAD constexpr).
// ---------------------------------------------------------------------------
template<int KPAD, int NFRAG, bool RELU>
__global__ __launch_bounds__(256) void mnf_gemm(
    const _Float16* __restrict__ Wt, const float* __restrict__ bias,
    const _Float16* __restrict__ X, int ldx,
    _Float16* __restrict__ Y, int ldy)
{
    __shared__ __align__(16) _Float16 xs[64 * KPAD];
    const int tid  = threadIdx.x;
    const int lane = tid & 31;
    const int wave = tid >> 5;
    const int mbase = blockIdx.x * 64;

    // ---- stage 64 x KPAD fp16 tile of X into LDS (b128 vectors) ----
    constexpr int VPR = KPAD / 8;             // 8-f16 vectors per row
    for (int i = tid; i < 64 * VPR; i += 256) {
        int r = i / VPR, c = (i - r * VPR) * 8;
        *(v8h*)(&xs[r * KPAD + c]) = *(const v8h*)(&X[(size_t)(mbase + r) * ldx + c]);
    }
    __syncthreads();

    v8f acc[4][NFRAG];
    #pragma unroll
    for (int mf = 0; mf < 4; ++mf)
        #pragma unroll
        for (int nf = 0; nf < NFRAG; ++nf)
            #pragma unroll
            for (int v = 0; v < 8; ++v) acc[mf][nf][v] = 0.0f;

    const int nlane  = lane & 15;
    const int halfA  = (lane < 16) ? 0 : 8;    // A: K-offset of low half per lane group
    const int halfB  = (lane < 16) ? 0 : 8;    // B: K-offset of low half per lane group
    const int nwbase = wave * (16 * NFRAG);

    #pragma unroll
    for (int k = 0; k < KPAD; k += 32) {
        // B fragments: lane holds column n, K = {k+halfB..+8, k+16+halfB..+8}
        v16h bfrag[NFRAG];
        #pragma unroll
        for (int nf = 0; nf < NFRAG; ++nf) {
            int n = nwbase + nf * 16 + nlane;
            const _Float16* wp = Wt + (size_t)n * KPAD + k + halfB;
            v8h blo = *(const v8h*)(wp);
            v8h bhi = *(const v8h*)(wp + 16);
            #pragma unroll
            for (int i = 0; i < 8; ++i) { bfrag[nf][i] = blo[i]; bfrag[nf][8 + i] = bhi[i]; }
        }
        #pragma unroll
        for (int mf = 0; mf < 4; ++mf) {
            // A fragment: row m = mf*16 + lane%16; K = {k+halfA..+8, k+16+halfA..+8}
            const _Float16* lp = &xs[(mf * 16 + nlane) * KPAD + k + halfA];
            v8h alo = *(const v8h*)(lp);
            v8h ahi = *(const v8h*)(lp + 16);
            v16h afrag;
            #pragma unroll
            for (int i = 0; i < 8; ++i) { afrag[i] = alo[i]; afrag[8 + i] = ahi[i]; }
            #pragma unroll
            for (int nf = 0; nf < NFRAG; ++nf)
                acc[mf][nf] = __builtin_amdgcn_wmma_f32_16x16x32_f16(
                    false, afrag, false, bfrag[nf], (short)0, acc[mf][nf], false, false);
        }
    }

    // ---- epilogue: bias (+ReLU), fp16 store.  D layout: n=lane%16, m=v+8*(lane/16)
    #pragma unroll
    for (int nf = 0; nf < NFRAG; ++nf) {
        int n = nwbase + nf * 16 + nlane;
        float bv = bias[n];
        #pragma unroll
        for (int mf = 0; mf < 4; ++mf) {
            #pragma unroll
            for (int v = 0; v < 8; ++v) {
                int row = mbase + mf * 16 + v + 8 * (lane >> 4);
                float y = acc[mf][nf][v] + bv;
                if (RELU) y = y > 0.0f ? y : 0.0f;
                Y[(size_t)row * ldy + n] = (_Float16)y;
            }
        }
    }
}

// ---------------------------------------------------------------------------
// Density (softplus) + RGB (sigmoid) heads, per sample
// ---------------------------------------------------------------------------
__global__ void mnf_heads(const _Float16* __restrict__ Xfinal /*[CH][352], x in 0..255*/,
                          const _Float16* __restrict__ H /*[CH][128]*/,
                          const float* __restrict__ w_density, const float* __restrict__ b_density,
                          const float* __restrict__ w_rgb, const float* __restrict__ b_rgb,
                          float* __restrict__ density, float* __restrict__ rgb, int mbase) {
    int r = blockIdx.x * blockDim.x + threadIdx.x;
    if (r >= CH) return;
    int m = mbase + r;
    const _Float16* x = Xfinal + (size_t)r * LDACT;
    float sd = 0.0f;
    for (int j = 0; j < 256; ++j) sd += (float)x[j] * w_density[j];
    float raw = sd + b_density[0] - 1.0f;               // DENSITY_BIAS = -1
    density[m] = (raw > 20.0f) ? raw : log1pf(expf(raw));

    const _Float16* h = H + (size_t)r * 128;
    #pragma unroll
    for (int c = 0; c < 3; ++c) {
        float s = 0.0f;
        for (int j = 0; j < 128; ++j) s += (float)h[j] * w_rgb[j * 3 + c];
        s += b_rgb[c];
        float sig = 1.0f / (1.0f + expf(-s));
        rgb[(size_t)m * 3 + c] = sig * 1.002f - 0.001f;  // RGB_PADDING = 0.001
    }
}

// ---------------------------------------------------------------------------
// Alpha compositing + volumetric rendering, one thread per ray
// ---------------------------------------------------------------------------
__global__ void mnf_render(const float* __restrict__ tvals, const float* __restrict__ dirs,
                           const float* __restrict__ density, const float* __restrict__ rgb,
                           float* __restrict__ wout, float* __restrict__ out, int level) {
    int b = blockIdx.x * blockDim.x + threadIdx.x;
    if (b >= NB) return;
    const float* tv = tvals + b * (NS + 1);
    float dn = sqrtf(dirs[b * 3] * dirs[b * 3] + dirs[b * 3 + 1] * dirs[b * 3 + 1] +
                     dirs[b * 3 + 2] * dirs[b * 3 + 2]);
    float T = 1.0f, accw = 0.0f, dsum = 0.0f;
    float cr0 = 0.0f, cr1 = 0.0f, cr2 = 0.0f;
    for (int s = 0; s < NS; ++s) {
        float dd = density[b * NS + s] * (tv[s + 1] - tv[s]) * dn;
        float ea = expf(-dd);
        float w = (1.0f - ea) * T;
        T *= ea;
        wout[b * NS + s] = w;
        accw += w;
        cr0 += w * rgb[(size_t)(b * NS + s) * 3 + 0];
        cr1 += w * rgb[(size_t)(b * NS + s) * 3 + 1];
        cr2 += w * rgb[(size_t)(b * NS + s) * 3 + 2];
        dsum += w * 0.5f * (tv[s] + tv[s + 1]);
    }
    float bg = 1.0f - accw;                               // WHITE_BG
    float dist = dsum / fmaxf(accw, 1e-10f);
    dist = fminf(fmaxf(dist, tv[0]), tv[NS]);
    float* o = out + b * 10 + level * 5;
    o[0] = cr0 + bg; o[1] = cr1 + bg; o[2] = cr2 + bg; o[3] = dist; o[4] = accw;
}

// ---------------------------------------------------------------------------
// Inverse-CDF resampling of t_vals (level 1), one thread per ray
// ---------------------------------------------------------------------------
__global__ void mnf_resample(float* __restrict__ tvals, const float* __restrict__ wsrc_all) {
    int b = blockIdx.x * blockDim.x + threadIdx.x;
    if (b >= NB) return;
    const float* w = wsrc_all + b * NS;
    float* tv = tvals + b * (NS + 1);

    auto wp = [&](int i) -> float {           // wp[130] = [w0, w..., w127]
        int j = i - 1; j = j < 0 ? 0 : (j > 127 ? 127 : j);
        return w[j];
    };
    float wpr[NS];
    float wsum = 0.0f;
    for (int i = 0; i < NS; ++i) {
        float wm0 = fmaxf(wp(i), wp(i + 1));
        float wm1 = fmaxf(wp(i + 1), wp(i + 2));
        wpr[i] = 0.5f * (wm0 + wm1) + 0.01f;  // RESAMPLE_PADDING
        wsum += wpr[i];
    }
    float pad = fmaxf(0.0f, 1e-5f - wsum);
    float dsum = wsum + pad;
    float padN = pad / (float)NS;

    float cdf[NS + 1];
    cdf[0] = 0.0f;
    float c = 0.0f;
    for (int i = 0; i < NS - 1; ++i) {        // cumsum(pdf[:-1])
        c += (wpr[i] + padN) / dsum;
        cdf[i + 1] = fminf(1.0f, c);
    }
    cdf[NS] = 1.0f;

    float nt[NS + 1];
    const float ueps = 1.1920929e-7f;
    for (int i = 0; i <= NS; ++i) {
        float u = (1.0f - ueps) * ((float)i / (float)NS);
        int idx = 0;
        for (int j = 0; j <= NS; ++j) idx += (cdf[j] <= u) ? 1 : 0;  // searchsorted right
        int below = idx - 1; below = below < 0 ? 0 : (below > NS ? NS : below);
        int above = idx;     above = above < 0 ? 0 : (above > NS ? NS : above);
        float g0 = cdf[below], g1 = cdf[above];
        float b0 = tv[below], b1 = tv[above];
        float dn2 = g1 - g0;
        float t = (dn2 > 0.0f) ? (u - g0) / dn2 : 0.0f;
        t = fminf(fmaxf(t, 0.0f), 1.0f);
        nt[i] = b0 + t * (b1 - b0);
    }
    for (int i = 0; i <= NS; ++i) tv[i] = nt[i];
}

// ---------------------------------------------------------------------------
// Host orchestration
// ---------------------------------------------------------------------------
extern "C" void kernel_launch(void* const* d_in, const int* in_sizes, int n_in,
                              void* d_out, int out_size, void* d_ws, size_t ws_size,
                              hipStream_t stream) {
    (void)in_sizes; (void)n_in; (void)out_size; (void)ws_size;
    const float* origins    = (const float*)d_in[0];
    const float* directions = (const float*)d_in[1];
    const float* viewdirs   = (const float*)d_in[2];
    const float* radii      = (const float*)d_in[3];
    const float* nearp      = (const float*)d_in[4];
    const float* farp       = (const float*)d_in[5];
    const float* wlay[8]; const float* blay[8];
    for (int i = 0; i < 8; ++i) { wlay[i] = (const float*)d_in[6 + 2 * i]; blay[i] = (const float*)d_in[7 + 2 * i]; }
    const float* w_density = (const float*)d_in[22]; const float* b_density = (const float*)d_in[23];
    const float* w_bott    = (const float*)d_in[24]; const float* b_bott    = (const float*)d_in[25];
    const float* w_view    = (const float*)d_in[26]; const float* b_view    = (const float*)d_in[27];
    const float* w_rgb     = (const float*)d_in[28]; const float* b_rgb     = (const float*)d_in[29];
    float* out = (float*)d_out;

    // ---- workspace layout ----
    char* ws = (char*)d_ws;
    size_t off = 0;
    auto take = [&](size_t bytes) -> void* {
        void* p = ws + off;
        off = (off + bytes + 255) & ~(size_t)255;
        return p;
    };
    static const int KPADL[8] = {96, 256, 256, 256, 256, 352, 256, 256};
    _Float16* wt[8];
    for (int i = 0; i < 8; ++i) wt[i] = (_Float16*)take((size_t)256 * KPADL[i] * 2);
    _Float16* wtB = (_Float16*)take((size_t)256 * 256 * 2);
    _Float16* wtV = (_Float16*)take((size_t)128 * 288 * 2);
    _Float16* Abuf = (_Float16*)take((size_t)CH * LDACT * 2);
    _Float16* Bbuf = (_Float16*)take((size_t)CH * LDACT * 2);
    _Float16* Hbuf = (_Float16*)take((size_t)CH * 128 * 2);
    float* density = (float*)take((size_t)MTOT * 4);
    float* rgbbuf  = (float*)take((size_t)MTOT * 3 * 4);
    float* tvals   = (float*)take((size_t)NB * (NS + 1) * 4);
    float* wbuf    = (float*)take((size_t)NB * NS * 4);

    // ---- one-time weight transposes (fp32 -> fp16 [N][KPAD]) ----
    auto tgrid = [](int n) { return dim3((unsigned)((n + 255) / 256)); };
    for (int i = 0; i < 8; ++i)
        mnf_transpose_w<<<tgrid(256 * KPADL[i]), 256, 0, stream>>>(wlay[i], wt[i], KPADL[i], 256, KPADL[i]);
    mnf_transpose_w<<<tgrid(256 * 256), 256, 0, stream>>>(w_bott, wtB, 256, 256, 256);
    mnf_transpose_w<<<tgrid(128 * 288), 256, 0, stream>>>(w_view, wtV, 283, 128, 288);

    mnf_tvals_init<<<tgrid(NB * (NS + 1)), 256, 0, stream>>>(nearp, farp, tvals);

    const dim3 gGemm(CH / 64), bGemm(256);
    const dim3 gRow(CH / 256), bRow(256);

    for (int level = 0; level < 2; ++level) {
        if (level == 1)
            mnf_resample<<<dim3(NB / 256), 256, 0, stream>>>(tvals, wbuf);

        for (int ch = 0; ch < NCHUNK; ++ch) {
            int mbase = ch * CH;
            mnf_prep<<<gRow, bRow, 0, stream>>>(tvals, origins, directions, radii, Abuf, Bbuf, mbase);
            // MLP trunk (skip-concat handled by IPE living at cols 256..351)
            mnf_gemm<96, 2, true ><<<gGemm, bGemm, 0, stream>>>(wt[0], blay[0], Abuf + 256, LDACT, Bbuf, LDACT);
            mnf_gemm<256, 2, true ><<<gGemm, bGemm, 0, stream>>>(wt[1], blay[1], Bbuf, LDACT, Abuf, LDACT);
            mnf_gemm<256, 2, true ><<<gGemm, bGemm, 0, stream>>>(wt[2], blay[2], Abuf, LDACT, Bbuf, LDACT);
            mnf_gemm<256, 2, true ><<<gGemm, bGemm, 0, stream>>>(wt[3], blay[3], Bbuf, LDACT, Abuf, LDACT);
            mnf_gemm<256, 2, true ><<<gGemm, bGemm, 0, stream>>>(wt[4], blay[4], Abuf, LDACT, Bbuf, LDACT);
            mnf_gemm<352, 2, true ><<<gGemm, bGemm, 0, stream>>>(wt[5], blay[5], Bbuf, LDACT, Abuf, LDACT);
            mnf_gemm<256, 2, true ><<<gGemm, bGemm, 0, stream>>>(wt[6], blay[6], Abuf, LDACT, Bbuf, LDACT);
            mnf_gemm<256, 2, true ><<<gGemm, bGemm, 0, stream>>>(wt[7], blay[7], Bbuf, LDACT, Abuf, LDACT);
            // view encoding into Bbuf cols 256..287, then heads
            mnf_vdenc<<<gRow, bRow, 0, stream>>>(viewdirs, Bbuf, mbase);
            mnf_gemm<256, 2, false><<<gGemm, bGemm, 0, stream>>>(wtB, b_bott, Abuf, LDACT, Bbuf, LDACT);
            mnf_gemm<288, 1, true ><<<gGemm, bGemm, 0, stream>>>(wtV, b_view, Bbuf, LDACT, Hbuf, 128);
            mnf_heads<<<gRow, bRow, 0, stream>>>(Abuf, Hbuf, w_density, b_density, w_rgb, b_rgb,
                                                 density, rgbbuf, mbase);
        }
        mnf_render<<<dim3(NB / 256), 256, 0, stream>>>(tvals, directions, density, rgbbuf,
                                                       wbuf, out, level);
    }
}
